// AttentionHeadVMP_51015621542085
// MI455X (gfx1250) — matrix operand order repaired and checked by reading, not compiled, not run on hardware
//
#include <hip/hip_runtime.h>
#include <hip/hip_bf16.h>

// MI455X / gfx1250, wave32. Moment-propagating attention (mean + variance).
// Strategy: all matmuls via v_wmma_f32_16x16x32_bf16; attention fused
// (flash-style two-pass) so the 512MB [B,H,S,S] logit tensors never hit HBM.
// Workspace requirement: ~114 MiB of bf16 staging (see offsets below).

typedef __attribute__((ext_vector_type(16))) __bf16 v16bf;
typedef __attribute__((ext_vector_type(8)))  __bf16 v8bf;
typedef __attribute__((ext_vector_type(8)))  float  v8f;

#define NTOK 4096   // B*S
#define DM   1024
#define SEQ  1024
#define NH   16
#define DH   64

// ---- workspace layout (bf16 element offsets) ----
static constexpr size_t MEGE    = 1u << 20;        // 1M elements
static constexpr size_t OFF_XB  = 0;               // bf16(x)          [4096,1024]
static constexpr size_t OFF_VXB = 4 * MEGE;        // bf16(var_x)
static constexpr size_t OFF_XSB = 8 * MEGE;        // bf16(x^2+var_x)
static constexpr size_t OFF_WB  = 12 * MEGE;       // per weight w: +w*3M: W, W*W, var_W
static constexpr size_t OFF_Q   = 21 * MEGE;       // q   natural [tok, d]
static constexpr size_t OFF_VQ  = 25 * MEGE;       // vq
static constexpr size_t OFF_Q2  = 29 * MEGE;       // q*q
static constexpr size_t OFF_K   = 33 * MEGE;       // k
static constexpr size_t OFF_VK  = 37 * MEGE;       // vk
static constexpr size_t OFF_K2  = 41 * MEGE;       // k*k
static constexpr size_t OFF_VT  = 45 * MEGE;       // v    transposed [B,H,dh,S]
static constexpr size_t OFF_VVT = 49 * MEGE;       // vv   transposed
static constexpr size_t OFF_W2T = 53 * MEGE;       // vv+v*v transposed
// total = 57M bf16 elements = 114 MiB

// ---------------------------------------------------------------------------
// Fragment helpers (CDNA5 wave32 WMMA VGPR layouts, CDNA5 ISA)
// ---------------------------------------------------------------------------
__device__ __forceinline__ v16bf mk16(v8bf lo, v8bf hi) {
  v16bf r;
#pragma unroll
  for (int i = 0; i < 8; ++i) { r[i] = lo[i]; r[i + 8] = hi[i]; }
  return r;
}

// A-matrix 16x32 bf16: lanes 0-15 row=lane, k={0..7,16..23};
//                      lanes 16-31 row=lane-16, k={8..15,24..31}
__device__ __forceinline__ v16bf load_a_frag(const __bf16* base, int ld, int lane) {
  int r = lane & 15, half = lane >> 4;
  const __bf16* p = base + (size_t)r * ld + half * 8;
  v8bf lo = *(const v8bf*)(p);
  v8bf hi = *(const v8bf*)(p + 16);
  return mk16(lo, hi);
}

// B-matrix 32x16 bf16 from "BT" storage (BT[n*ld + k]):
// lanes 0-15: n=lane, k={0..7,16..23} ; lanes 16-31: n=lane-16, k={8..15,24..31}
__device__ __forceinline__ v16bf load_b_frag(const __bf16* baseT, int ld, int lane) {
  int n = lane & 15, half = lane >> 4;
  const __bf16* p = baseT + (size_t)n * ld + half * 8;
  v8bf lo = *(const v8bf*)(p);
  v8bf hi = *(const v8bf*)(p + 16);
  return mk16(lo, hi);
}

__device__ __forceinline__ v8f wmma_bf16(v16bf a, v16bf b, v8f c) {
  return __builtin_amdgcn_wmma_f32_16x16x32_bf16(false, a, false, b, (short)0, c,
                                                 false, false);
}

// ---------------------------------------------------------------------------
// Prep kernels: f32 -> bf16 operand staging
// ---------------------------------------------------------------------------
__global__ __launch_bounds__(256) void prep_inputs_kernel(
    const float* __restrict__ x, const float* __restrict__ vx, __bf16* ws) {
  size_t i = (size_t)blockIdx.x * 256 + threadIdx.x;
  if (i >= (size_t)NTOK * DM) return;
  float xv = x[i], vv = vx[i];
  ws[OFF_XB + i]  = (__bf16)xv;
  ws[OFF_VXB + i] = (__bf16)vv;
  ws[OFF_XSB + i] = (__bf16)(xv * xv + vv);
}

__global__ __launch_bounds__(256) void prep_weights_kernel(
    const float* __restrict__ Wq, const float* __restrict__ vWq,
    const float* __restrict__ Wk, const float* __restrict__ vWk,
    const float* __restrict__ Wv, const float* __restrict__ vWv, __bf16* ws) {
  int w = blockIdx.z;
  const float* W  = (w == 0) ? Wq : (w == 1) ? Wk : Wv;
  const float* vW = (w == 0) ? vWq : (w == 1) ? vWk : vWv;
  size_t i = (size_t)blockIdx.x * 256 + threadIdx.x;
  if (i >= (size_t)DM * DM) return;
  size_t base = OFF_WB + (size_t)w * 3 * MEGE;
  float wv = W[i];
  ws[base + i]            = (__bf16)wv;
  ws[base + MEGE + i]     = (__bf16)(wv * wv);
  ws[base + 2 * MEGE + i] = (__bf16)vW[i];
}

// ---------------------------------------------------------------------------
// Projection GEMMs:  m = x @ W^T ; v = var_x @ (W*W)^T + (x^2+var_x) @ var_W^T
// Each wave computes a 16x32 output tile; 6 wmma per 32-wide k-step.
// ---------------------------------------------------------------------------
__device__ __forceinline__ void gemm_tile(const __bf16* ws, int widx, int m0, int n0,
                                          int lane, v8f accM[2], v8f accV[2]) {
  const __bf16* A0 = ws + OFF_XB;
  const __bf16* A1 = ws + OFF_VXB;
  const __bf16* A2 = ws + OFF_XSB;
  size_t wb = OFF_WB + (size_t)widx * 3 * MEGE;
  const __bf16* B0 = ws + wb;             // W      (BT layout: [n,k])
  const __bf16* B1 = ws + wb + MEGE;      // W*W
  const __bf16* B2 = ws + wb + 2 * MEGE;  // var_W
  for (int kk = 0; kk < DM; kk += 32) {
    if (kk + 32 < DM)  // hint next B panel into cache
      __builtin_prefetch(B0 + (size_t)n0 * DM + kk + 32, 0, 1);
    v16bf aX = load_a_frag(A0 + (size_t)m0 * DM + kk, DM, lane);
    v16bf aV = load_a_frag(A1 + (size_t)m0 * DM + kk, DM, lane);
    v16bf aS = load_a_frag(A2 + (size_t)m0 * DM + kk, DM, lane);
#pragma unroll
    for (int t = 0; t < 2; ++t) {
      size_t bo = (size_t)(n0 + t * 16) * DM + kk;
      v16bf bW = load_b_frag(B0 + bo, DM, lane);
      v16bf bS = load_b_frag(B1 + bo, DM, lane);
      v16bf bV = load_b_frag(B2 + bo, DM, lane);
      accM[t] = wmma_bf16(aX, bW, accM[t]);
      accV[t] = wmma_bf16(aV, bS, accV[t]);
      accV[t] = wmma_bf16(aS, bV, accV[t]);
    }
  }
}

// widx 0/1 (Q/K): natural [tok, d] output of (m, v, m*m)
__global__ __launch_bounds__(256) void gemm_qk_kernel(__bf16* ws) {
  int widx = blockIdx.z;  // 0=Q, 1=K
  int wgl  = blockIdx.x * 8 + (threadIdx.x >> 5);  // 0..8191
  int lane = threadIdx.x & 31;
  int m0 = (wgl >> 5) * 16;
  int n0 = (wgl & 31) * 32;
  v8f accM[2] = {}, accV[2] = {};
  gemm_tile(ws, widx, m0, n0, lane, accM, accV);
  __bf16* oM = ws + (widx ? OFF_K : OFF_Q);
  __bf16* oV = ws + (widx ? OFF_VK : OFF_VQ);
  __bf16* o2 = ws + (widx ? OFF_K2 : OFF_Q2);
  int ln = lane & 15, half = lane >> 4;
#pragma unroll
  for (int t = 0; t < 2; ++t)
#pragma unroll
    for (int r = 0; r < 8; ++r) {
      int row = m0 + r + half * 8;
      int col = n0 + t * 16 + ln;
      size_t gi = (size_t)row * DM + col;
      float m = accM[t][r], v = accV[t][r];
      oM[gi] = (__bf16)m;
      oV[gi] = (__bf16)v;
      o2[gi] = (__bf16)(m * m);
    }
}

// widx 2 (V): transposed [B,H,dh,S] output of (v, vv, vv+v*v)
__global__ __launch_bounds__(256) void gemm_v_kernel(__bf16* ws) {
  int wgl  = blockIdx.x * 8 + (threadIdx.x >> 5);
  int lane = threadIdx.x & 31;
  int m0 = (wgl >> 5) * 16;
  int n0 = (wgl & 31) * 32;
  v8f accM[2] = {}, accV[2] = {};
  gemm_tile(ws, 2, m0, n0, lane, accM, accV);
  __bf16* oV  = ws + OFF_VT;
  __bf16* oVV = ws + OFF_VVT;
  __bf16* oW2 = ws + OFF_W2T;
  int ln = lane & 15, half = lane >> 4;
  int b = m0 >> 10;
#pragma unroll
  for (int t = 0; t < 2; ++t)
#pragma unroll
    for (int r = 0; r < 8; ++r) {
      int s   = (m0 & (SEQ - 1)) + r + half * 8;
      int col = n0 + t * 16 + ln;           // col in [0,1024): h=col>>6, dh=col&63
      int h = col >> 6, dh = col & 63;
      size_t gi = (((size_t)(b * NH + h) * DH + dh) << 10) + s;
      float m = accM[t][r], v = accV[t][r];
      oV[gi]  = (__bf16)m;
      oVV[gi] = (__bf16)v;
      oW2[gi] = (__bf16)(v + m * m);
    }
}

// ---------------------------------------------------------------------------
// Fused variance-propagating attention + residual.
// Block = 8 waves; each wave owns 16 query rows of one (b,h).
// Pass A: online rowmax / sum-exp of logits a = qk^T/32.
// Pass B: recompute a and va = (vq@vk^T + vq@(k^2)^T + q^2@vk^T)/1024,
//   s = softmax, g = s^2(1-2s)va, then WMMA-accumulate
//   o = s@v, U = g@(vv+v^2) + s^2@vv, T2 = s^2@(vv+v^2), sv = sum s^2 va.
//   vo = U + sv * T2.
// ---------------------------------------------------------------------------
__global__ __launch_bounds__(256) void attn_vmp_kernel(
    const __bf16* __restrict__ ws, const float* __restrict__ x,
    const float* __restrict__ vx, float* __restrict__ out) {
  __shared__ __bf16 lds_buf[8 * 3 * 512];  // per-wave 3x (16x32) bf16 tiles

  int bh = blockIdx.x;
  int b = bh >> 4, h = bh & 15;
  int wave = threadIdx.x >> 5, lane = threadIdx.x & 31;
  int ln = lane & 15, half = lane >> 4;
  int s0 = blockIdx.y * 128 + wave * 16;

  const __bf16* qb  = ws + OFF_Q  + ((size_t)(b * SEQ + s0)) * DM + h * DH;
  const __bf16* vqb = ws + OFF_VQ + ((size_t)(b * SEQ + s0)) * DM + h * DH;
  const __bf16* q2b = ws + OFF_Q2 + ((size_t)(b * SEQ + s0)) * DM + h * DH;
  const __bf16* kb  = ws + OFF_K  + ((size_t)(b * SEQ)) * DM + h * DH;
  const __bf16* vkb = ws + OFF_VK + ((size_t)(b * SEQ)) * DM + h * DH;
  const __bf16* k2b = ws + OFF_K2 + ((size_t)(b * SEQ)) * DM + h * DH;
  const __bf16* vtb  = ws + OFF_VT  + ((size_t)(b * NH + h) * DH) * SEQ;
  const __bf16* vvtb = ws + OFF_VVT + ((size_t)(b * NH + h) * DH) * SEQ;
  const __bf16* w2tb = ws + OFF_W2T + ((size_t)(b * NH + h) * DH) * SEQ;

  // persistent A fragments: q, vq, q^2 for k-steps 0/1 (dh 0..31, 32..63)
  v16bf aq[2], avq[2], aq2[2];
#pragma unroll
  for (int ks = 0; ks < 2; ++ks) {
    aq[ks]  = load_a_frag(qb  + ks * 32, DM, lane);
    avq[ks] = load_a_frag(vqb + ks * 32, DM, lane);
    aq2[ks] = load_a_frag(q2b + ks * 32, DM, lane);
  }

  const float inv32 = 0.03125f;        // 1/sqrt(D)
  const float invD  = 1.0f / 1024.0f;  // 1/D

  // ---- pass A: softmax stats ----
  float rowm[8], rowl[8];
#pragma unroll
  for (int r = 0; r < 8; ++r) { rowm[r] = -1e30f; rowl[r] = 0.0f; }

  for (int kb0 = 0; kb0 < SEQ; kb0 += 32) {
    v8f a[2];
#pragma unroll
    for (int t = 0; t < 2; ++t) {
      const __bf16* base = kb + (size_t)(kb0 + t * 16) * DM;
      v8f acc = {};
      acc = wmma_bf16(aq[0], load_b_frag(base, DM, lane), acc);
      acc = wmma_bf16(aq[1], load_b_frag(base + 32, DM, lane), acc);
      a[t] = acc;
    }
#pragma unroll
    for (int r = 0; r < 8; ++r) {
      float tm = fmaxf(a[0][r], a[1][r]) * inv32;
#pragma unroll
      for (int m = 8; m >= 1; m >>= 1) tm = fmaxf(tm, __shfl_xor(tm, m, 32));
      float newm = fmaxf(rowm[r], tm);
      float p = __expf(a[0][r] * inv32 - newm) + __expf(a[1][r] * inv32 - newm);
#pragma unroll
      for (int m = 8; m >= 1; m >>= 1) p += __shfl_xor(p, m, 32);
      rowl[r] = rowl[r] * __expf(rowm[r] - newm) + p;
      rowm[r] = newm;
    }
  }
  float rowinv[8];
#pragma unroll
  for (int r = 0; r < 8; ++r) rowinv[r] = 1.0f / rowl[r];

  // ---- pass B: recompute + accumulate ----
  __bf16* myS  = lds_buf + wave * 3 * 512;
  __bf16* myG  = myS + 512;
  __bf16* myS2 = myS + 1024;

  v8f o[4] = {}, U[4] = {}, T2[4] = {};
  float svp[8] = {};

  for (int kb0 = 0; kb0 < SEQ; kb0 += 32) {
    v8f a[2], va[2];
#pragma unroll
    for (int t = 0; t < 2; ++t) {
      size_t ko = (size_t)(kb0 + t * 16) * DM;
      v16bf bk0  = load_b_frag(kb  + ko, DM, lane);
      v16bf bk1  = load_b_frag(kb  + ko + 32, DM, lane);
      v16bf bvk0 = load_b_frag(vkb + ko, DM, lane);
      v16bf bvk1 = load_b_frag(vkb + ko + 32, DM, lane);
      v16bf bk20 = load_b_frag(k2b + ko, DM, lane);
      v16bf bk21 = load_b_frag(k2b + ko + 32, DM, lane);
      v8f am = {};
      am = wmma_bf16(aq[0], bk0, am);
      am = wmma_bf16(aq[1], bk1, am);
      a[t] = am;
      v8f av = {};
      av = wmma_bf16(avq[0], bvk0, av);
      av = wmma_bf16(avq[1], bvk1, av);
      av = wmma_bf16(avq[0], bk20, av);
      av = wmma_bf16(avq[1], bk21, av);
      av = wmma_bf16(aq2[0], bvk0, av);
      av = wmma_bf16(aq2[1], bvk1, av);
      va[t] = av;
    }
    // softmax delta-method pieces, spilled to LDS for C->A layout transpose
#pragma unroll
    for (int t = 0; t < 2; ++t)
#pragma unroll
      for (int r = 0; r < 8; ++r) {
        float vA = va[t][r] * invD;
        float s  = __expf(a[t][r] * inv32 - rowm[r]) * rowinv[r];
        float s2 = s * s;
        float g  = s2 * (1.0f - 2.0f * s) * vA;
        svp[r] += s2 * vA;
        int idx = (r + half * 8) * 32 + t * 16 + ln;
        myS[idx]  = (__bf16)s;
        myG[idx]  = (__bf16)g;
        myS2[idx] = (__bf16)s2;
      }
    asm volatile("s_wait_dscnt 0" ::: "memory");
    v16bf as_ = load_a_frag(myS, 32, lane);
    v16bf ag  = load_a_frag(myG, 32, lane);
    v16bf as2 = load_a_frag(myS2, 32, lane);
#pragma unroll
    for (int t4 = 0; t4 < 4; ++t4) {
      size_t vo_ = (size_t)(t4 * 16) * SEQ + kb0;
      v16bf bv  = load_b_frag(vtb  + vo_, SEQ, lane);
      v16bf bvv = load_b_frag(vvtb + vo_, SEQ, lane);
      v16bf bw2 = load_b_frag(w2tb + vo_, SEQ, lane);
      o[t4]  = wmma_bf16(as_, bv, o[t4]);
      U[t4]  = wmma_bf16(ag, bw2, U[t4]);
      U[t4]  = wmma_bf16(as2, bvv, U[t4]);
      T2[t4] = wmma_bf16(as2, bw2, T2[t4]);
    }
  }

  // row-wise sv reduction across the 16-lane group
#pragma unroll
  for (int r = 0; r < 8; ++r)
#pragma unroll
    for (int m = 8; m >= 1; m >>= 1) svp[r] += __shfl_xor(svp[r], m, 32);

  // epilogue: residual add, write mean + variance halves of d_out
  float* outM = out;
  float* outV = out + (size_t)NTOK * DM;
#pragma unroll
  for (int t4 = 0; t4 < 4; ++t4)
#pragma unroll
    for (int r = 0; r < 8; ++r) {
      int row = s0 + r + half * 8;
      int col = h * DH + t4 * 16 + ln;
      size_t gi = (size_t)(b * SEQ + row) * DM + col;
      outM[gi] = x[gi] + o[t4][r];
      outV[gi] = vx[gi] + U[t4][r] + svp[r] * T2[t4][r];
    }
}

// ---------------------------------------------------------------------------
extern "C" void kernel_launch(void* const* d_in, const int* in_sizes, int n_in,
                              void* d_out, int out_size, void* d_ws, size_t ws_size,
                              hipStream_t stream) {
  const float* x   = (const float*)d_in[0];
  const float* vx  = (const float*)d_in[1];
  const float* Wq  = (const float*)d_in[2];
  const float* vWq = (const float*)d_in[3];
  const float* Wk  = (const float*)d_in[4];
  const float* vWk = (const float*)d_in[5];
  const float* Wv  = (const float*)d_in[6];
  const float* vWv = (const float*)d_in[7];
  __bf16* ws = (__bf16*)d_ws;
  float* out = (float*)d_out;

  prep_inputs_kernel<<<dim3(16384), dim3(256), 0, stream>>>(x, vx, ws);
  prep_weights_kernel<<<dim3(4096, 1, 3), dim3(256), 0, stream>>>(Wq, vWq, Wk, vWk,
                                                                  Wv, vWv, ws);
  gemm_qk_kernel<<<dim3(1024, 1, 2), dim3(256), 0, stream>>>(ws);
  gemm_v_kernel<<<dim3(1024, 1, 1), dim3(256), 0, stream>>>(ws);
  attn_vmp_kernel<<<dim3(64, 8), dim3(256), 0, stream>>>(ws, x, vx, out);
}
